// IDW_impute_low_high_pass_62268435857709
// MI455X (gfx1250) — matrix operand order repaired and checked by reading, not compiled, not run on hardware
//
#include <hip/hip_runtime.h>
#include <hip/hip_bf16.h>

typedef __attribute__((ext_vector_type(16))) _Float16 v16h;
typedef __attribute__((ext_vector_type(8)))  float    v8f;

typedef unsigned int uint32x4 __attribute__((ext_vector_type(4)));
typedef int          int32x8  __attribute__((ext_vector_type(8)));

#define BATCH 32
#define TMAX  512
#define DIMM  128

__device__ __forceinline__ float softplusf(float x) {
    return logf(1.0f + expf(x));
}

// ---------------- ws layout (bytes) ----------------
// table1 : [128][512] f16   at 0          (131072)
// table2 : [128][512] f16   at 131072     (131072)
// coeffH : [128][128] f16   at 262144     (32768)
// Apack  : [128][64][512] f16 at 294912   (8388608)   rows 0..31 = masked_data, 32..63 = mask
// Aeach  : [32*512][128] f16 at 8683520   (4194304)   infl_each in (b,t,d) order for GEMM2
#define WS_TAB1   0
#define WS_TAB2   131072
#define WS_COEFF  262144
#define WS_APACK  294912
#define WS_AEACH  8683520

// ---- Tensor Data Mover: inline asm (portable across both toolchains; bypasses the
// ---- clang builtin whose arity differs). 1-D f16 strip global -> LDS.
// ---- D# per CDNA5 ISA; 2-group form (VADDR2/3 NULL, tensor <= 2D).
__device__ __forceinline__ void tdm_load_f16_1d(unsigned lds_off, const _Float16* gsrc, int nelem) {
    unsigned long long ga = (unsigned long long)(uintptr_t)gsrc;
    uint32x4 g0;
    g0[0] = 1u;                                                // count=1, user descriptor
    g0[1] = lds_off;                                           // lds_addr (bytes)
    g0[2] = (unsigned)(ga & 0xffffffffu);                      // global_addr[31:0]
    g0[3] = (unsigned)((ga >> 32) & 0x01ffffffu) | (2u << 30); // global_addr[56:32] | type=2
    int32x8 g1;
    g1[0] = 0x00010000;                                        // wg_mask=0, data_size=1 (2 bytes)
    g1[1] = (nelem & 0xffff) << 16;                            // tensor_dim0[15:0]
    g1[2] = ((nelem >> 16) & 0xffff) | (1 << 16);              // tensor_dim0[31:16], tensor_dim1=1
    g1[3] = (nelem & 0xffff) << 16;                            // tile_dim0 = nelem
    g1[4] = 0;                                                 // tile_dim1/2 unused
    g1[5] = nelem;                                             // tensor_dim0_stride[31:0]
    g1[6] = 0;
    g1[7] = 0;
    asm volatile("tensor_load_to_lds %0, %1" :: "s"(g0), "s"(g1) : "memory");
}

// ---------------- prep: per-d |s-t| kernel tables ----------------
__global__ void prep_tables(const float* __restrict__ lgw, const float* __restrict__ kk,
                            _Float16* __restrict__ t1, _Float16* __restrict__ t2) {
    int d = blockIdx.x;
    float w1 = softplusf(lgw[d]);
    float w2 = softplusf(kk[d]) + 1.0f;
    for (int i = threadIdx.x; i < TMAX; i += blockDim.x) {
        float q = (float)(i * i) * 0.01f;      // ((s-t)/10)^2
        t1[d * TMAX + i] = (_Float16)expf(-q * w1);
        t2[d * TMAX + i] = (_Float16)expf(-q * w2);
    }
}

__global__ void prep_coeff(const float* __restrict__ lcdc, _Float16* __restrict__ coeffH) {
    int i = blockIdx.x * blockDim.x + threadIdx.x;   // < 128*128
    coeffH[i] = (_Float16)softplusf(lcdc[i]);
}

// pack [masked_data ; mask] as Apack[d][row][s] f16
__global__ void prep_apack(const float* __restrict__ data, const float* __restrict__ mask,
                           _Float16* __restrict__ Apack) {
    int i = blockIdx.x * blockDim.x + threadIdx.x;   // < 128*64*512, d fastest for coalesced reads
    int d   = i & 127;
    int r   = i >> 7;
    int s   = r & (TMAX - 1);
    int row = r >> 9;                                 // 0..63
    int b   = row & 31;
    size_t gi = ((size_t)(b * TMAX + s)) * DIMM + d;
    float mk = mask[gi];
    float v  = (row < 32) ? data[gi] * mk : mk;
    Apack[((size_t)(d * 64 + row)) * TMAX + s] = (_Float16)v;
}

// ---------------- GEMM1: per-d (64x512)x(512x512) with WMMA ----------------
// grid.x = 128*8 : d = bx>>3 , tchunk = bx&7 (64 t-values per block)
// 8 waves: kern = w>>2 (K1/K2), pairIdx = (w&3)>>1, ntHalf = (w&3)&1
__global__ void __launch_bounds__(256)
gemm1_kernel(const _Float16* __restrict__ Apack,
             const _Float16* __restrict__ tab1, const _Float16* __restrict__ tab2,
             float* __restrict__ out1,      // influence_agg_weights (b,d,t)
             float* __restrict__ out3,      // high_pass pre-pass: each2/agg2 (b,d,t)
             _Float16* __restrict__ Aeach)  // infl_each as (b,t,d) f16
{
    __shared__ __align__(16) _Float16 tab[2][TMAX];

    const int d      = blockIdx.x >> 3;
    const int tchunk = blockIdx.x & 7;

    // Stage the two 1 KB kernel tables through the Tensor Data Mover (wave 0 only;
    // TDM issue ignores EXEC, the branch keeps other waves from duplicating the DMA).
    if (threadIdx.x < 32) {
        tdm_load_f16_1d((unsigned)(uintptr_t)&tab[0][0], tab1 + (size_t)d * TMAX, TMAX);
        tdm_load_f16_1d((unsigned)(uintptr_t)&tab[1][0], tab2 + (size_t)d * TMAX, TMAX);
        __builtin_amdgcn_s_wait_tensorcnt(0);
    }
    __syncthreads();

    const int w    = threadIdx.x >> 5;
    const int lane = threadIdx.x & 31;
    const int g    = lane >> 4;          // half-wave
    const int ln   = lane & 15;

    const int kern     = w >> 2;
    const int pairIdx  = (w & 3) >> 1;
    const int ntHalf   = (w & 3) & 1;
    const int pairBase = pairIdx * 16;   // rows for "each" tile; "+32" rows for "agg"

    const _Float16* tp = tab[kern];
    const _Float16* Ad = Apack + ((size_t)d * 64) * TMAX;

    for (int nt = 0; nt < 2; ++nt) {
        const int tBase = tchunk * 64 + (ntHalf * 2 + nt) * 16;
        const int tcol  = tBase + ln;

        v8f accE = {};
        v8f accA = {};

        for (int ks = 0; ks < 16; ++ks) {
            const int sBase = ks * 32;

            // B fragment: lane holds column t=tcol, k = h + 16*g
            union { v16h v; _Float16 h[16]; } B;
            const int s0 = sBase + 16 * g;
#pragma unroll
            for (int h = 0; h < 16; ++h) {
                int dlt = s0 + h - tcol;
                if (dlt < 0) dlt = -dlt;
                B.h[h] = tp[dlt];
            }

            // A fragments (pair pattern: kb = 16g + 2*p)
            union { v16h v; unsigned u[8]; } AE, AA;
            const _Float16* rowE = Ad + ((size_t)(pairBase + ln)) * TMAX + sBase;
            const _Float16* rowA = Ad + ((size_t)(pairBase + 32 + ln)) * TMAX + sBase;
            __builtin_prefetch(rowE + 32, 0, 0);   // global_prefetch_b8: next k-chunk
            __builtin_prefetch(rowA + 32, 0, 0);
#pragma unroll
            for (int p = 0; p < 8; ++p) {
                const int kb = g * 16 + 2 * p;
                AE.u[p] = *reinterpret_cast<const unsigned*>(rowE + kb);
                AA.u[p] = *reinterpret_cast<const unsigned*>(rowA + kb);
            }

            accE = __builtin_amdgcn_wmma_f32_16x16x32_f16(
                false, AE.v, false, B.v, (short)0, accE, false, false);
            accA = __builtin_amdgcn_wmma_f32_16x16x32_f16(
                false, AA.v, false, B.v, (short)0, accA, false, false);
        }

        // C/D layout: VGPR v -> row = base + v + 8*g, col = tcol
        if (kern == 0) {
#pragma unroll
            for (int v = 0; v < 8; ++v) {
                const int b = pairBase + v + 8 * g;   // batch index for both tiles
                Aeach[((size_t)(b * TMAX + tcol)) * DIMM + d] = (_Float16)accE[v];
                out1[((size_t)(b * DIMM + d)) * TMAX + tcol]  = accA[v];
            }
        } else {
#pragma unroll
            for (int v = 0; v < 8; ++v) {
                const int b = pairBase + v + 8 * g;
                // denominators are kernel-mass sums (>= O(1)); v_rcp_f32 is plenty accurate
                out3[((size_t)(b * DIMM + d)) * TMAX + tcol] =
                    accE[v] * __builtin_amdgcn_rcpf(accA[v]);
            }
        }
    }
}

// ---------------- cross_dims: sum over d of out1 ----------------
__global__ void reduce_cross(const float* __restrict__ out1, float* __restrict__ out5) {
    int i = blockIdx.x * blockDim.x + threadIdx.x;  // < 32*512
    int b = i >> 9;
    int t = i & (TMAX - 1);
    float s = 0.0f;
    const float* p = out1 + (size_t)b * DIMM * TMAX + t;
#pragma unroll 4
    for (int d = 0; d < DIMM; ++d) s += p[(size_t)d * TMAX];
    out5[i] = s;
}

// ---------------- GEMM2: agg_influence = (b*T x d) * coeff(d x j) ----------------
// grid.x = 128 blocks, 8 waves, wave owns one 16-row m-tile, loops 8 n-tiles x 4 k-steps
__global__ void __launch_bounds__(256)
gemm2_kernel(const _Float16* __restrict__ Aeach, const _Float16* __restrict__ coeffH,
             float* __restrict__ out4) {
    __shared__ __align__(16) _Float16 Bs[DIMM * DIMM];   // 32 KB
    for (int i = threadIdx.x; i < DIMM * DIMM; i += blockDim.x) Bs[i] = coeffH[i];
    __syncthreads();

    const int w    = threadIdx.x >> 5;
    const int lane = threadIdx.x & 31;
    const int g    = lane >> 4;
    const int ln   = lane & 15;

    const int rowBase = (blockIdx.x * 8 + w) * 16;   // bt rows

    for (int nt = 0; nt < 8; ++nt) {
        v8f acc = {};
        const int ncol = nt * 16 + ln;
#pragma unroll
        for (int ks = 0; ks < 4; ++ks) {
            union { v16h v; unsigned u[8]; } A;
            const _Float16* rowp = Aeach + ((size_t)(rowBase + ln)) * DIMM + ks * 32;
#pragma unroll
            for (int p = 0; p < 8; ++p) {
                const int kb = g * 16 + 2 * p;
                A.u[p] = *reinterpret_cast<const unsigned*>(rowp + kb);
            }
            union { v16h v; _Float16 h[16]; } B;
#pragma unroll
            for (int h = 0; h < 16; ++h) {
                const int k = ks * 32 + 16 * g + h;
                B.h[h] = Bs[k * DIMM + ncol];
            }
            acc = __builtin_amdgcn_wmma_f32_16x16x32_f16(
                false, A.v, false, B.v, (short)0, acc, false, false);
        }
#pragma unroll
        for (int v = 0; v < 8; ++v) {
            const int r = rowBase + v + 8 * g;
            out4[(size_t)r * DIMM + ncol] = acc[v];
        }
    }
}

// ---------------- finalize: low_pass + high_pass ----------------
__global__ void finalize_kernel(const float* __restrict__ out4, const float* __restrict__ out5,
                                float* __restrict__ out2, float* __restrict__ out3) {
    int i  = blockIdx.x * blockDim.x + threadIdx.x;  // < 32*512*128  (b,t,j)
    int bt = i >> 7;
    int j  = i & 127;
    float lp = out4[i] * __builtin_amdgcn_rcpf(out5[bt]);  // cross_dims is a large positive sum
    out2[i] = lp;
    int b = bt >> 9;
    int t = bt & (TMAX - 1);
    size_t o3 = ((size_t)(b * DIMM + j)) * TMAX + t;
    out3[o3] = out3[o3] - lp;
}

extern "C" void kernel_launch(void* const* d_in, const int* in_sizes, int n_in,
                              void* d_out, int out_size, void* d_ws, size_t ws_size,
                              hipStream_t stream) {
    const float* data = (const float*)d_in[0];
    const float* mask = (const float*)d_in[1];
    // d_in[2] = T_max (fixed at 512)
    const float* kk   = (const float*)d_in[3];
    const float* lgw  = (const float*)d_in[4];
    const float* lcdc = (const float*)d_in[5];

    char* ws = (char*)d_ws;
    _Float16* tab1   = (_Float16*)(ws + WS_TAB1);
    _Float16* tab2   = (_Float16*)(ws + WS_TAB2);
    _Float16* coeffH = (_Float16*)(ws + WS_COEFF);
    _Float16* Apack  = (_Float16*)(ws + WS_APACK);
    _Float16* Aeach  = (_Float16*)(ws + WS_AEACH);

    float* out  = (float*)d_out;
    float* out1 = out;                 // influence_agg_weights (32,128,512)
    float* out2 = out + 2097152;       // low_pass_res          (32,512,128)
    float* out3 = out + 4194304;       // high_pass_res         (32,128,512)
    float* out4 = out + 6291456;       // agg_influence         (32,512,128)
    float* out5 = out + 8388608;       // cross_dims_bt1        (32,512,1)

    prep_tables<<<DIMM, 256, 0, stream>>>(lgw, kk, tab1, tab2);
    prep_coeff<<<(DIMM * DIMM) / 256, 256, 0, stream>>>(lcdc, coeffH);
    prep_apack<<<(DIMM * 64 * TMAX) / 256, 256, 0, stream>>>(data, mask, Apack);

    gemm1_kernel<<<DIMM * 8, 256, 0, stream>>>(Apack, tab1, tab2, out1, out3, Aeach);

    reduce_cross<<<(BATCH * TMAX) / 256, 256, 0, stream>>>(out1, out5);

    gemm2_kernel<<<(BATCH * TMAX) / (8 * 16), 256, 0, stream>>>(Aeach, coeffH, out4);

    finalize_kernel<<<(BATCH * TMAX * DIMM) / 256, 256, 0, stream>>>(out4, out5, out2, out3);
}
